// T5GNNAdapt_50276887167422
// MI455X (gfx1250) — matrix operand order repaired and checked by reading, not compiled, not run on hardware
//
#include <hip/hip_runtime.h>
#include <hip/hip_bf16.h>

// ---------------------------------------------------------------------------
// T5 GNN adapter block for MI455X (gfx1250, wave32, WMMA).
//
// Pipeline:
//   k0: memset agg/cnt (hipMemsetAsync)
//   k1: LayerNorm -> x_f32 [N,D] and x_bf16 [N,D]
//   k2: LDS-tiled transpose+f32->bf16 of weight[2] (D,F)->(F,D),
//       root (D,F)->(F,D), wo (F,D)->(D,F)   (K-contiguous B layout for WMMA)
//   k3: edge scatter: global_atomic_add_f32 into agg[2,N,D], cnt[2,N]
//   k4: mean + bf16 -> h_bf16 [2,N,D]
//   k5: GEMM1 (bf16 WMMA): act = relu(x@rootT + h0@W0T + h1@W1T + bias) -> bf16 [N,F]
//   k6: GEMM2 (bf16 WMMA): out = hidden + act@woT -> f32 [N,D]
//
// GEMM tiling: 64x64 per wave (4x4 WMMA accumulators), 8 waves/block (2Mx4N)
// -> 128x256 block tile. Fragment loads use uniform-base + 32-bit lane offset
// (GVS addressing) so the K-loop advances with a single scalar add.
// ---------------------------------------------------------------------------

#define NN   8192      // B*S nodes
#define DD   1024
#define FF   4096
#define NE   262144
#define LN_EPS 1e-6f

typedef __attribute__((ext_vector_type(16))) __bf16 v16bf;
typedef __attribute__((ext_vector_type(8)))  __bf16 v8bf;
typedef __attribute__((ext_vector_type(8)))  float  v8f;

__device__ __forceinline__ unsigned short f2bf(float f) {
    unsigned int u = __float_as_uint(f);
    u += 0x7FFFu + ((u >> 16) & 1u);     // round-to-nearest-even
    return (unsigned short)(u >> 16);
}

// ------------------------- k1: LayerNorm -----------------------------------
__global__ void ln_kernel(const float* __restrict__ hs,
                          const float* __restrict__ gamma,
                          const float* __restrict__ beta,
                          float* __restrict__ x, unsigned short* __restrict__ xb) {
    const int row = blockIdx.x;            // 8192 rows
    const int t   = threadIdx.x;           // 256 threads, 4 elems each
    const float4 v = reinterpret_cast<const float4*>(hs + (size_t)row * DD)[t];

    __shared__ float s1[256], s2[256];
    s1[t] = v.x + v.y + v.z + v.w;
    s2[t] = v.x*v.x + v.y*v.y + v.z*v.z + v.w*v.w;
    __syncthreads();
    for (int s = 128; s > 0; s >>= 1) {
        if (t < s) { s1[t] += s1[t + s]; s2[t] += s2[t + s]; }
        __syncthreads();
    }
    const float mu  = s1[0] * (1.0f / DD);
    const float var = s2[0] * (1.0f / DD) - mu * mu;
    const float rs  = rsqrtf(var + LN_EPS);

    const int j = t * 4;
    const float4 g = reinterpret_cast<const float4*>(gamma)[t];
    const float4 b = reinterpret_cast<const float4*>(beta)[t];
    float4 o;
    o.x = (v.x - mu) * rs * g.x + b.x;
    o.y = (v.y - mu) * rs * g.y + b.y;
    o.z = (v.z - mu) * rs * g.z + b.z;
    o.w = (v.w - mu) * rs * g.w + b.w;
    reinterpret_cast<float4*>(x + (size_t)row * DD)[t] = o;
    unsigned short* xo = xb + (size_t)row * DD + j;
    xo[0] = f2bf(o.x); xo[1] = f2bf(o.y); xo[2] = f2bf(o.z); xo[3] = f2bf(o.w);
}

// --------------- k2: transpose + convert f32[R,C] -> bf16[C,R] --------------
__global__ void tconv_kernel(const float* __restrict__ in,
                             unsigned short* __restrict__ out, int R, int C) {
    __shared__ float tile[32][33];
    const int bx = blockIdx.x * 32;                // col base (C)
    const int by = blockIdx.y * 32;                // row base (R)
    const int tx = threadIdx.x & 31;
    const int ty = threadIdx.x >> 5;               // 8 rows per step
    #pragma unroll
    for (int i = 0; i < 32; i += 8)
        tile[ty + i][tx] = in[(size_t)(by + ty + i) * C + (bx + tx)];
    __syncthreads();
    #pragma unroll
    for (int i = 0; i < 32; i += 8)
        out[(size_t)(bx + ty + i) * R + (by + tx)] = f2bf(tile[tx][ty + i]);
}

// ------------------------- k3: edge scatter ---------------------------------
__global__ void edge_kernel(const float* __restrict__ x,
                            const int* __restrict__ src,
                            const int* __restrict__ dst,
                            const int* __restrict__ etype,
                            float* __restrict__ agg, float* __restrict__ cnt) {
    const int e = blockIdx.x;
    const int s = src[e], d = dst[e], r = etype[e];
    const int j = threadIdx.x * 4;
    const float4 v = *reinterpret_cast<const float4*>(x + (size_t)s * DD + j);
    float* a = agg + ((size_t)r * NN + d) * DD + j;
    atomicAdd(a + 0, v.x);
    atomicAdd(a + 1, v.y);
    atomicAdd(a + 2, v.z);
    atomicAdd(a + 3, v.w);
    if (threadIdx.x == 0) atomicAdd(cnt + (size_t)r * NN + d, 1.0f);
}

// ------------------------- k4: mean + bf16 ----------------------------------
__global__ void mean_kernel(const float* __restrict__ agg,
                            const float* __restrict__ cnt,
                            unsigned short* __restrict__ h) {
    const size_t idx = (size_t)blockIdx.x * blockDim.x + threadIdx.x; // 2*N*D
    const float c = cnt[idx >> 10];
    h[idx] = f2bf(agg[idx] * (1.0f / fmaxf(c, 1.0f)));
}

// -------------------- WMMA fragment loaders (ISA 7.12.2) --------------------
// A [M,K] row-major bf16: lane lm row, two 8-elem (16B) chunks at +0 / +16.
__device__ __forceinline__ v16bf load_a(const unsigned short* base, unsigned off) {
    union { v16bf v; v8bf h[2]; } u;
    u.h[0] = *reinterpret_cast<const v8bf*>(base + off);
    u.h[1] = *reinterpret_cast<const v8bf*>(base + off + 16);
    return u.v;
}
// B transposed [Nmat,K]: lane ln col, two 8-elem (16B) chunks at +0 / +16.
__device__ __forceinline__ v16bf load_b(const unsigned short* base, unsigned off) {
    union { v16bf v; v8bf h[2]; } u;
    u.h[0] = *reinterpret_cast<const v8bf*>(base + off);
    u.h[1] = *reinterpret_cast<const v8bf*>(base + off + 16);
    return u.v;
}

// ----- k5: GEMM1  act[N,F] = relu(x@rootT + h0@W0T + h1@W1T + bias) ---------
// 256 threads = 8 waves; waves 2(M)x4(N); wave tile 64x64; block tile 128x256.
__global__ void __launch_bounds__(256)
gemm1_kernel(const unsigned short* __restrict__ xb,
             const unsigned short* __restrict__ h0,
             const unsigned short* __restrict__ h1,
             const unsigned short* __restrict__ rootT,
             const unsigned short* __restrict__ w0T,
             const unsigned short* __restrict__ w1T,
             const float* __restrict__ bias,
             unsigned short* __restrict__ act) {
    const int tid  = threadIdx.x;
    const int wave = tid >> 5, lane = tid & 31;
    const int lm = lane & 15, lh = lane >> 4;
    const int m0 = blockIdx.y * 128 + (wave >> 2) * 64;
    const int n0 = blockIdx.x * 256 + (wave & 3) * 64;

    // 32-bit per-lane element offsets (constant over K loop; K=DD for all pairs)
    unsigned aoff[4], boff[4];
    #pragma unroll
    for (int i = 0; i < 4; ++i) aoff[i] = (unsigned)(m0 + 16 * i + lm) * DD + lh * 8;
    #pragma unroll
    for (int j = 0; j < 4; ++j) boff[j] = (unsigned)(n0 + 16 * j + lm) * DD + lh * 8;

    v8f acc[4][4] = {};

    const unsigned short* As[3] = { xb, h0, h1 };
    const unsigned short* Bs[3] = { rootT, w0T, w1T };
    #pragma unroll
    for (int p = 0; p < 3; ++p) {
        const unsigned short* A = As[p];   // uniform; advances by scalar add
        const unsigned short* B = Bs[p];
        #pragma unroll 2
        for (int k0 = 0; k0 < DD; k0 += 32) {
            v16bf af[4], bf[4];
            #pragma unroll
            for (int i = 0; i < 4; ++i) af[i] = load_a(A, aoff[i]);
            #pragma unroll
            for (int j = 0; j < 4; ++j) bf[j] = load_b(B, boff[j]);
            // speculative prefetch ~4 K-steps ahead (reuses base+voffset operands)
            __builtin_prefetch(A + aoff[0] + 128, 0, 1);
            __builtin_prefetch(B + boff[0] + 128, 0, 1);
            #pragma unroll
            for (int i = 0; i < 4; ++i)
                #pragma unroll
                for (int j = 0; j < 4; ++j)
                    acc[i][j] = __builtin_amdgcn_wmma_f32_16x16x32_bf16(
                        false, af[i], false, bf[j], (short)0, acc[i][j], false, false);
            A += 32;
            B += 32;
        }
    }

    float bcol[4];
    #pragma unroll
    for (int j = 0; j < 4; ++j) bcol[j] = bias[n0 + 16 * j + lm];
    #pragma unroll
    for (int i = 0; i < 4; ++i) {
        #pragma unroll
        for (int ii = 0; ii < 8; ++ii) {     // D layout: row = base + lh*8 + ii
            const int r = m0 + 16 * i + lh * 8 + ii;
            #pragma unroll
            for (int j = 0; j < 4; ++j) {
                const int c = n0 + 16 * j + lm;
                act[(size_t)r * FF + c] = f2bf(fmaxf(acc[i][j][ii] + bcol[j], 0.0f));
            }
        }
    }
}

// ----- k6: GEMM2  out[N,D] = hidden + act@woT ------------------------------
__global__ void __launch_bounds__(256)
gemm2_kernel(const unsigned short* __restrict__ act,
             const unsigned short* __restrict__ woT,
             const float* __restrict__ hidden,
             float* __restrict__ out) {
    const int tid  = threadIdx.x;
    const int wave = tid >> 5, lane = tid & 31;
    const int lm = lane & 15, lh = lane >> 4;
    const int m0 = blockIdx.y * 128 + (wave >> 2) * 64;
    const int n0 = blockIdx.x * 256 + (wave & 3) * 64;

    unsigned aoff[4], boff[4];
    #pragma unroll
    for (int i = 0; i < 4; ++i) aoff[i] = (unsigned)(m0 + 16 * i + lm) * FF + lh * 8;
    #pragma unroll
    for (int j = 0; j < 4; ++j) boff[j] = (unsigned)(n0 + 16 * j + lm) * FF + lh * 8;

    v8f acc[4][4] = {};
    const unsigned short* A = act;
    const unsigned short* B = woT;
    #pragma unroll 2
    for (int k0 = 0; k0 < FF; k0 += 32) {
        v16bf af[4], bf[4];
        #pragma unroll
        for (int i = 0; i < 4; ++i) af[i] = load_a(A, aoff[i]);
        #pragma unroll
        for (int j = 0; j < 4; ++j) bf[j] = load_b(B, boff[j]);
        __builtin_prefetch(A + aoff[0] + 128, 0, 1);
        __builtin_prefetch(B + boff[0] + 128, 0, 1);
        #pragma unroll
        for (int i = 0; i < 4; ++i)
            #pragma unroll
            for (int j = 0; j < 4; ++j)
                acc[i][j] = __builtin_amdgcn_wmma_f32_16x16x32_bf16(
                    false, af[i], false, bf[j], (short)0, acc[i][j], false, false);
        A += 32;
        B += 32;
    }

    #pragma unroll
    for (int i = 0; i < 4; ++i) {
        #pragma unroll
        for (int ii = 0; ii < 8; ++ii) {
            const int r = m0 + 16 * i + lh * 8 + ii;
            #pragma unroll
            for (int j = 0; j < 4; ++j) {
                const int c = n0 + 16 * j + lm;
                out[(size_t)r * DD + c] = hidden[(size_t)r * DD + c] + acc[i][j][ii];
            }
        }
    }
}

// ---------------------------------------------------------------------------
extern "C" void kernel_launch(void* const* d_in, const int* in_sizes, int n_in,
                              void* d_out, int out_size, void* d_ws, size_t ws_size,
                              hipStream_t stream) {
    const float* hidden = (const float*)d_in[0];        // [8,1024,1024]
    const float* weight = (const float*)d_in[1];        // [2,1024,4096]
    const float* root   = (const float*)d_in[2];        // [1024,4096]
    const float* bias   = (const float*)d_in[3];        // [4096]
    const float* wo     = (const float*)d_in[4];        // [4096,1024]
    const float* gamma  = (const float*)d_in[5];        // [1024]
    const float* beta   = (const float*)d_in[6];        // [1024]
    const int*   eidx   = (const int*)d_in[7];          // [2,262144]
    const int*   etype  = (const int*)d_in[8];          // [262144]
    float* out = (float*)d_out;

    // workspace carve (all offsets 256B-aligned)
    char* ws = (char*)d_ws;
    size_t off = 0;
    float*          x_f32 = (float*)(ws + off);          off += (size_t)NN * DD * 4;
    unsigned short* x_bf  = (unsigned short*)(ws + off); off += (size_t)NN * DD * 2;
    unsigned short* w0T   = (unsigned short*)(ws + off); off += (size_t)FF * DD * 2;
    unsigned short* w1T   = (unsigned short*)(ws + off); off += (size_t)FF * DD * 2;
    unsigned short* rootT = (unsigned short*)(ws + off); off += (size_t)FF * DD * 2;
    unsigned short* woT   = (unsigned short*)(ws + off); off += (size_t)DD * FF * 2;
    float*          agg   = (float*)(ws + off);          off += (size_t)2 * NN * DD * 4;
    float*          cnt   = (float*)(ws + off);          off += (size_t)2 * NN * 4;
    unsigned short* h_bf  = (unsigned short*)(ws + off); off += (size_t)2 * NN * DD * 2;
    unsigned short* act   = (unsigned short*)(ws + off); off += (size_t)NN * FF * 2;
    (void)ws_size; (void)off; (void)in_sizes; (void)n_in; (void)out_size;

    hipMemsetAsync(agg, 0, (size_t)2 * NN * DD * 4, stream);
    hipMemsetAsync(cnt, 0, (size_t)2 * NN * 4, stream);

    ln_kernel<<<NN, 256, 0, stream>>>(hidden, gamma, beta, x_f32, x_bf);

    // weight[r] (D,F) -> (F,D);  root (D,F) -> (F,D);  wo (F,D) -> (D,F)
    dim3 gW(FF / 32, DD / 32);
    tconv_kernel<<<gW, 256, 0, stream>>>(weight,                          w0T, DD, FF);
    tconv_kernel<<<gW, 256, 0, stream>>>(weight + (size_t)DD * FF,        w1T, DD, FF);
    tconv_kernel<<<gW, 256, 0, stream>>>(root,                            rootT, DD, FF);
    dim3 gWo(DD / 32, FF / 32);
    tconv_kernel<<<gWo, 256, 0, stream>>>(wo,                             woT, FF, DD);

    edge_kernel<<<NE, 256, 0, stream>>>(x_f32, eidx, eidx + NE, etype, agg, cnt);

    mean_kernel<<<(2u * NN * DD) / 256, 256, 0, stream>>>(agg, cnt, h_bf);

    dim3 g1(FF / 256, NN / 128);
    gemm1_kernel<<<g1, 256, 0, stream>>>(x_bf, h_bf, h_bf + (size_t)NN * DD,
                                         rootT, w0T, w1T, bias, act);

    dim3 g2(DD / 256, NN / 128);
    gemm2_kernel<<<g2, 256, 0, stream>>>(act, woT, hidden, out);
}
